// VectorQuantizer_45500883534477
// MI455X (gfx1250) — matrix operand order repaired and checked by reading, not compiled, not run on hardware
//
#include <hip/hip_runtime.h>

// ---------------------------------------------------------------------------
// VQ-VAE vector quantizer for MI455X (gfx1250, wave32, WMMA).
//
// argmin_k dist[n][k],  dist = ||z_n||^2 + ||e_k||^2 - 2<z_n,e_k>
// Cross term: 131072x1024x64 GEMM -> v_wmma_f32_16x16x32_f16 (f32 accum).
// ||z||^2 is constant per row -> removed from the argmin compare path and
// re-added only for the loss. The streaming argmin uses a packed u32 key:
//   key = (bits(en + 1 - 2*dot) & ~0x3FF) | code
// which is monotonic (biased distance > 0) -> one v_fma + v_and_or_b32 +
// v_min_u32 per element, tie-break to lowest code for free.
//
// Block: 256 threads (8 wave32), 256 z-rows vs all 1024 codes. Codebook is
// staged per block into LDS as f16 (row stride 66 -> 33-bank stride kills the
// 16-lane column-read conflicts). HBM traffic ~68 MB; GEMM 17.2 GFLOP f16.
// ---------------------------------------------------------------------------

typedef __attribute__((ext_vector_type(16))) _Float16 v16h;
typedef __attribute__((ext_vector_type(8)))  float    v8f;

#define DIMS            64
#define NUM_EMB         1024
#define N_ROWS          131072          // 4 * 32 * 32 * 32
#define ZQ_ELEMS        8388608         // N_ROWS * DIMS
#define ROWS_PER_BLOCK  256
#define SPATIAL         32768           // 32*32*32 (stride of channel dim)
#define ESTRIDE         66

// LDS carve (bytes), all 16B aligned
#define OFF_ESM    0                    // _Float16 [1024*66]       = 135168 B
#define OFF_ZSM    135168               // _Float16 [256*66]        =  33792 B
#define OFF_ENORM  168960               // float   [1024]           =   4096 B
#define OFF_ZNORM  173056               // float   [256]            =   1024 B
#define OFF_BVAL   174080               // float   [256]            =   1024 B
#define OFF_BIDX   175104               // int     [256]            =   1024 B
#define OFF_WSUM   176128               // float   [8]              =     32 B
#define SMEM_BYTES 176160               // < 320 KB per WGP

union HF { v16h h; unsigned int u[8]; };

__global__ void vq_wmma_kernel(const float* __restrict__ z,
                               const float* __restrict__ emb,
                               float* __restrict__ out)
{
    extern __shared__ char smem[];
    _Float16* esm   = (_Float16*)(smem + OFF_ESM);
    _Float16* zsm   = (_Float16*)(smem + OFF_ZSM);
    float*    enorm = (float*)   (smem + OFF_ENORM);
    float*    znorm = (float*)   (smem + OFF_ZNORM);
    float*    bval  = (float*)   (smem + OFF_BVAL);
    int*      bidx  = (int*)     (smem + OFF_BIDX);
    float*    wsum  = (float*)   (smem + OFF_WSUM);

    const int t  = threadIdx.x;
    const int n0 = blockIdx.x * ROWS_PER_BLOCK;       // 32768 % 256 == 0
    const int b  = n0 >> 15;
    const int r0 = n0 & (SPATIAL - 1);
    const float* zb = z + (size_t)b * DIMS * SPATIAL;

    // ---- Stage z tile (f16) + exact f32 row norms. Coalesced 256-wide. ----
    {
        float acc = 0.0f;
        #pragma unroll 8
        for (int c = 0; c < DIMS; ++c) {
            float v = zb[(size_t)c * SPATIAL + r0 + t];
            acc = fmaf(v, v, acc);
            zsm[t * ESTRIDE + c] = (_Float16)v;
        }
        znorm[t] = acc;
    }
    // ---- Stage codebook (f16), coalesced 256 consecutive floats / iter ----
    for (int i = 0; i < (NUM_EMB * DIMS) / 256; ++i) {
        int e = i * 256 + t;
        esm[(e >> 6) * ESTRIDE + (e & 63)] = (_Float16)emb[e];
    }
    // ---- Exact f32 codebook norms (4 codes / thread, L2-hot) ----
    #pragma unroll
    for (int q = 0; q < 4; ++q) {
        int code = t * 4 + q;
        const float* er = emb + (size_t)code * DIMS;
        float s = 0.0f;
        #pragma unroll 8
        for (int c = 0; c < DIMS; ++c) s = fmaf(er[c], er[c], s);
        enorm[code] = s;
    }
    __syncthreads();

    // ---- Per-wave WMMA argmin: wave w owns rows [w*32, w*32+32) ----
    const int lane = t & 31;
    const int w    = t >> 5;
    const int mr   = lane & 15;     // A: row M; B/C/D: column N
    const int half = lane >> 4;

    // A fragments (16-bit A 16x32 ISA layout): VGPR v holds K-pair
    //   (v<4 ? 2v : 2v+8) + 8*half ; both lane halves hold M=0..15.
    HF a[2][2];
    #pragma unroll
    for (int t2 = 0; t2 < 2; ++t2) {
        int row = w * 32 + t2 * 16 + mr;
        #pragma unroll
        for (int kc = 0; kc < 2; ++kc) {
            #pragma unroll
            for (int v = 0; v < 8; ++v) {
                int kp = kc * 32 + ((v < 4) ? 2 * v : 2 * v + 8) + 8 * half;
                a[t2][kc].u[v] = *(const unsigned int*)(zsm + row * ESTRIDE + kp);
            }
        }
    }

    // Packed (biased-dist | code) keys; 0xFFFFFFFF > any real key (dist > 0).
    unsigned int key[2][8];
    #pragma unroll
    for (int t2 = 0; t2 < 2; ++t2)
        #pragma unroll
        for (int j = 0; j < 8; ++j) key[t2][j] = 0xFFFFFFFFu;

    for (int kb = 0; kb < NUM_EMB / 16; ++kb) {
        const int mycode = kb * 16 + mr;          // column this lane holds in C/D
        // B fragments (same K order as A): VGPR v holds K-pair
        //   (v<4 ? 2v : 2v+8) + 8*half.
        HF bm[2];
        #pragma unroll
        for (int kc = 0; kc < 2; ++kc) {
            #pragma unroll
            for (int v = 0; v < 8; ++v) {
                int kd = kc * 32 + ((v < 4) ? 2 * v : 2 * v + 8) + 8 * half;
                bm[kc].u[v] = *(const unsigned int*)(esm + mycode * ESTRIDE + kd);
            }
        }
        const float enb = enorm[mycode] + 1.0f;   // bias keeps key positive

        #pragma unroll
        for (int t2 = 0; t2 < 2; ++t2) {
            v8f cacc = {};
            cacc = __builtin_amdgcn_wmma_f32_16x16x32_f16(false, a[t2][0].h,
                                                          false, bm[0].h,
                                                          (short)0, cacc, false, false);
            cacc = __builtin_amdgcn_wmma_f32_16x16x32_f16(false, a[t2][1].h,
                                                          false, bm[1].h,
                                                          (short)0, cacc, false, false);
            #pragma unroll
            for (int j = 0; j < 8; ++j) {
                float biased = fmaf(-2.0f, cacc[j], enb);           // v_fma
                unsigned int cand =                                  // v_and_or_b32
                    (__float_as_uint(biased) & 0xFFFFFC00u) | (unsigned int)mycode;
                unsigned int k = key[t2][j];                         // v_min_u32
                key[t2][j] = (cand < k) ? cand : k;
            }
        }
    }

    // ---- Cross-lane min over the 16 columns holding each row ----
    #pragma unroll
    for (int t2 = 0; t2 < 2; ++t2) {
        #pragma unroll
        for (int j = 0; j < 8; ++j) {
            unsigned int k = key[t2][j];
            #pragma unroll
            for (int off = 8; off >= 1; off >>= 1) {
                unsigned int o = __shfl_xor(k, off, 32);
                k = (o < k) ? o : k;
            }
            if (mr == 0) {   // lanes 0 (rows 0..7 of tile) and 16 (rows 8..15)
                int row = w * 32 + t2 * 16 + 8 * half + j;
                bidx[row] = (int)(k & 1023u);
                // min dist = (biased - 1) + ||z||^2 ; truncation error ~2^-13
                bval[row] = __uint_as_float(k & 0xFFFFFC00u) - 1.0f + znorm[row];
            }
        }
    }
    __syncthreads();

    // ---- Outputs ----
    const int idx = bidx[t];
    // encoding_indices (third output), stored as float values
    out[ZQ_ELEMS + 1 + n0 + t] = (float)idx;

    // z_q = embedding[idx] in exact f32, [B,C,D,H,W] layout; consecutive
    // threads -> consecutive spatial addresses (coalesced stores).
    {
        const float* er = emb + (size_t)idx * DIMS;
        const size_t obase = (size_t)b * DIMS * SPATIAL + (size_t)(r0 + t);
        #pragma unroll 8
        for (int c = 0; c < DIMS; ++c)
            out[obase + (size_t)c * SPATIAL] = er[c];
    }

    // loss = 0.25 * mean(min_dist); min_dist == ||e_idx - z||^2
    float lv = fmaxf(bval[t], 0.0f);
    #pragma unroll
    for (int off = 16; off >= 1; off >>= 1) lv += __shfl_xor(lv, off, 32);
    if (lane == 0) wsum[w] = lv;
    __syncthreads();
    if (t == 0) {
        float s = 0.0f;
        #pragma unroll
        for (int i = 0; i < 8; ++i) s += wsum[i];
        atomicAdd(out + ZQ_ELEMS, s * (0.25f / (float)ZQ_ELEMS));
    }
}

extern "C" void kernel_launch(void* const* d_in, const int* in_sizes, int n_in,
                              void* d_out, int out_size, void* d_ws, size_t ws_size,
                              hipStream_t stream) {
    (void)in_sizes; (void)n_in; (void)out_size; (void)d_ws; (void)ws_size;
    const float* z   = (const float*)d_in[0];   // [4, 64, 32, 32, 32] f32
    const float* emb = (const float*)d_in[1];   // [1024, 64] f32
    float* out = (float*)d_out;                 // [zq | loss | indices]

    // Zero the loss accumulator slot (graph-capturable memset).
    hipMemsetAsync(out + ZQ_ELEMS, 0, sizeof(float), stream);

    dim3 grid(N_ROWS / ROWS_PER_BLOCK);         // 512 blocks
    dim3 block(ROWS_PER_BLOCK);                 // 256 threads = 8 wave32
    vq_wmma_kernel<<<grid, block, SMEM_BYTES, stream>>>(z, emb, out);
}
